// Attention_52295521796806
// MI455X (gfx1250) — matrix operand order repaired and checked by reading, not compiled, not run on hardware
//
#include <hip/hip_runtime.h>
#include <hip/hip_bf16.h>

// CDNA5 / gfx1250 fused windowed-attention (Swin) kernel.
// One workgroup (8 wave32) per window; all GEMMs via v_wmma_f32_16x16x32_f16.

typedef _Float16 v16h __attribute__((ext_vector_type(16)));
typedef _Float16 v8h  __attribute__((ext_vector_type(8)));
typedef float    v8f  __attribute__((ext_vector_type(8)));

#define TDIM   512
#define NHEADS 8
#define DHEAD  64
#define NTOK   25
#define MPAD   32
#define WMAT   (TDIM * TDIM)

static __device__ __forceinline__ v16h cat16(v8h lo, v8h hi) {
  v16h r;
#pragma unroll
  for (int i = 0; i < 8; ++i) { r[i] = lo[i]; r[i + 8] = hi[i]; }
  return r;
}

static __device__ __forceinline__ v8f wmma16(v16h a, v16h b, v8f c) {
  // (neg_a, A, neg_b, B, c_mod, C, reuse_a, reuse_b)
  return __builtin_amdgcn_wmma_f32_16x16x32_f16(false, a, false, b, (short)0, c,
                                                false, false);
}

// Pre-pack the four 512x512 f32 weight matrices into f16, transposed to
// [out][in] so WMMA B-operand reads are contiguous 16B runs per lane.
__global__ void pack_weights_kernel(const float* __restrict__ qw,
                                    const float* __restrict__ kw,
                                    const float* __restrict__ vw,
                                    const float* __restrict__ pw,
                                    _Float16* __restrict__ wpk) {
  int i = blockIdx.x * 256 + threadIdx.x;
  if (i >= WMAT) return;
  int r = i >> 9;              // input channel (row of [in][out])
  int c = i & 511;             // output channel
  int o = c * TDIM + r;        // transposed [out][in]
  wpk[0 * WMAT + o] = (_Float16)qw[i];
  wpk[1 * WMAT + o] = (_Float16)kw[i];
  wpk[2 * WMAT + o] = (_Float16)vw[i];
  wpk[3 * WMAT + o] = (_Float16)pw[i];
}

__global__ __launch_bounds__(256, 1)
void swin_window_attn_kernel(const float* __restrict__ x,
                             const float* __restrict__ qbias,
                             const float* __restrict__ kbias,
                             const float* __restrict__ vbias,
                             const float* __restrict__ pbias,
                             const float* __restrict__ rel_table,
                             const int*   __restrict__ rel_index,
                             const _Float16* __restrict__ wpk,
                             float* __restrict__ out) {
  // 128 KB static LDS (WGP has 320 KB). Score buffer aliases qs after QK^T;
  // output tile buffer aliases xs after QKV.
  __shared__ __align__(16) _Float16 xs[MPAD * TDIM];   // x, later attn output
  __shared__ __align__(16) _Float16 qs[MPAD * TDIM];   // Q, later f32 scores / f16 P
  __shared__ __align__(16) _Float16 ks[MPAD * TDIM];   // K row-major [tok][ch]
  __shared__ __align__(16) _Float16 vsT[TDIM * MPAD];  // V transposed [ch][tok]

  const int tid  = threadIdx.x;
  const int wave = tid >> 5;      // 8 waves == 8 heads
  const int lane = tid & 31;
  const int hs   = lane >> 4;     // half-wave select (WMMA layout)
  const int l15  = lane & 15;

  // ---------------- Phase 1: stage x window into LDS as f16 ----------------
  const float* __restrict__ xw = x + (size_t)blockIdx.x * NTOK * TDIM;
  for (int e = tid; e < MPAD * TDIM; e += 256) {
    int m = e >> 9, c = e & 511;
    float v = (m < NTOK) ? xw[m * TDIM + c] : 0.0f;
    xs[e] = (_Float16)v;
  }
  __syncthreads();

  const _Float16* wq = wpk;
  const _Float16* wk = wpk + WMAT;
  const _Float16* wv = wpk + 2 * WMAT;
  const _Float16* wp = wpk + 3 * WMAT;

  // ---------------- Phase 2: QKV projections (M=32, N=512, K=512) ----------
  // 192 output tiles (3 mats x 2 Mtiles x 32 Ntiles), 24 per wave, uniform.
  for (int t = wave; t < 192; t += NHEADS) {
    const int mat = t >> 6;
    const int r   = t & 63;
    const int m0  = (r >> 5) * 16;
    const int n0  = (r & 31) * 16;
    const _Float16* wT = (mat == 0) ? wq : (mat == 1) ? wk : wv;
    const float*    bv = (mat == 0) ? qbias : (mat == 1) ? kbias : vbias;
    const _Float16* xrow = &xs[(m0 + l15) * TDIM];
    const _Float16* wrow = &wT[(size_t)(n0 + l15) * TDIM];
    v8f acc = {};
#pragma unroll
    for (int kk = 0; kk < 16; ++kk) {
      const int k0 = kk * 32;
      v8h alo = *(const v8h*)(xrow + k0 + hs * 8);        // A: K 0..7 / 8..15
      v8h ahi = *(const v8h*)(xrow + k0 + 16 + hs * 8);   //    K 16..23 / 24..31
      v8h blo = *(const v8h*)(wrow + k0 + hs * 8);        // B: K 0..7 / 8..15
      v8h bhi = *(const v8h*)(wrow + k0 + 16 + hs * 8);   //    K 16..23 / 24..31
      acc = wmma16(cat16(alo, ahi), cat16(blo, bhi), acc);
    }
    const float bn = bv[n0 + l15];
    _Float16* dst = (mat == 0) ? qs : ks;
#pragma unroll
    for (int i = 0; i < 8; ++i) {
      const int m = m0 + hs * 8 + i;           // C layout: M = vgpr + 8*half
      const float v = acc[i] + bn;
      if (mat == 2) vsT[(n0 + l15) * MPAD + m] = (_Float16)v;   // V transposed
      else          dst[m * TDIM + n0 + l15]   = (_Float16)v;
    }
  }
  __syncthreads();

  // ---------------- Phase 3: attention, one wave per head ------------------
  const int h = wave;
  v8f s00 = {}, s01 = {}, s10 = {}, s11 = {};
#pragma unroll
  for (int kk = 0; kk < 2; ++kk) {
    const int k0 = h * DHEAD + kk * 32;
    const _Float16* q0 = &qs[(0 + l15) * TDIM];
    const _Float16* q1 = &qs[(16 + l15) * TDIM];
    v16h A0 = cat16(*(const v8h*)(q0 + k0 + hs * 8),
                    *(const v8h*)(q0 + k0 + 16 + hs * 8));
    v16h A1 = cat16(*(const v8h*)(q1 + k0 + hs * 8),
                    *(const v8h*)(q1 + k0 + 16 + hs * 8));
    const _Float16* kr0 = &ks[(0 + l15) * TDIM];
    const _Float16* kr1 = &ks[(16 + l15) * TDIM];
    v16h B0 = cat16(*(const v8h*)(kr0 + k0 + hs * 8),
                    *(const v8h*)(kr0 + k0 + 16 + hs * 8));
    v16h B1 = cat16(*(const v8h*)(kr1 + k0 + hs * 8),
                    *(const v8h*)(kr1 + k0 + 16 + hs * 8));
    s00 = wmma16(A0, B0, s00);
    s01 = wmma16(A0, B1, s01);
    s10 = wmma16(A1, B0, s10);
    s11 = wmma16(A1, B1, s11);
  }
  __syncthreads();  // everyone done reading qs -> safe to alias with scores

  float* sc = (float*)qs + h * (MPAD * MPAD);  // per-head 32x32 f32 scores
  auto emit_scores = [&](v8f s, int mt, int nt) {
#pragma unroll
    for (int i = 0; i < 8; ++i) {
      const int m = mt * 16 + hs * 8 + i;
      const int n = nt * 16 + l15;
      float v = s[i] * 0.125f;                 // dh^-0.5, dh=64
      if (n < NTOK) {
        if (m < NTOK)
          v += rel_table[(size_t)rel_index[m * NTOK + n] * NHEADS + h];
      } else {
        v = -1e30f;                            // mask padded key columns
      }
      sc[m * MPAD + n] = v;
    }
  };
  emit_scores(s00, 0, 0); emit_scores(s01, 0, 1);
  emit_scores(s10, 1, 0); emit_scores(s11, 1, 1);
  __syncthreads();

  // Softmax: one lane per row; write P back in-place as f16 (A-operand ready)
  {
    const int m = lane;
    float r[MPAD];
    float mx = -1e30f;
#pragma unroll
    for (int n = 0; n < MPAD; ++n) { r[n] = sc[m * MPAD + n]; mx = fmaxf(mx, r[n]); }
    float sum = 0.0f;
#pragma unroll
    for (int n = 0; n < MPAD; ++n) { float e = __expf(r[n] - mx); r[n] = e; sum += e; }
    const float inv = 1.0f / sum;
    _Float16* ph = (_Float16*)sc;
#pragma unroll
    for (int n = 0; n < MPAD; ++n) ph[m * MPAD + n] = (_Float16)(r[n] * inv);
  }
  __syncthreads();

  // attn @ V : [32x32] x [32x64], K=32 -> single WMMA per 16x16 tile
  const _Float16* ph = (const _Float16*)sc;
#pragma unroll
  for (int mt = 0; mt < 2; ++mt) {
    const _Float16* pr = ph + (mt * 16 + l15) * MPAD;
    v16h A = cat16(*(const v8h*)(pr + hs * 8),
                   *(const v8h*)(pr + 16 + hs * 8));
#pragma unroll
    for (int nt = 0; nt < 4; ++nt) {
      const _Float16* vr = &vsT[(h * DHEAD + nt * 16 + l15) * MPAD];
      v16h B = cat16(*(const v8h*)(vr + hs * 8),
                     *(const v8h*)(vr + 16 + hs * 8));
      v8f z = {};
      v8f o = wmma16(A, B, z);
#pragma unroll
      for (int i = 0; i < 8; ++i) {
        const int m = mt * 16 + hs * 8 + i;
        xs[m * TDIM + h * DHEAD + nt * 16 + l15] = (_Float16)o[i];  // reuse xs
      }
    }
  }
  __syncthreads();

  // ---------------- Phase 4: output projection + f32 store -----------------
  float* __restrict__ orow = out + (size_t)blockIdx.x * NTOK * TDIM;
  for (int t = wave; t < 64; t += NHEADS) {
    const int m0 = (t >> 5) * 16;
    const int n0 = (t & 31) * 16;
    const _Float16* xrow = &xs[(m0 + l15) * TDIM];
    const _Float16* wrow = &wp[(size_t)(n0 + l15) * TDIM];
    v8f acc = {};
#pragma unroll
    for (int kk = 0; kk < 16; ++kk) {
      const int k0 = kk * 32;
      v8h alo = *(const v8h*)(xrow + k0 + hs * 8);
      v8h ahi = *(const v8h*)(xrow + k0 + 16 + hs * 8);
      v8h blo = *(const v8h*)(wrow + k0 + hs * 8);
      v8h bhi = *(const v8h*)(wrow + k0 + 16 + hs * 8);
      acc = wmma16(cat16(alo, ahi), cat16(blo, bhi), acc);
    }
    const float bn = pbias[n0 + l15];
#pragma unroll
    for (int i = 0; i < 8; ++i) {
      const int m = m0 + hs * 8 + i;
      if (m < NTOK) orow[m * TDIM + n0 + l15] = acc[i] + bn;
    }
  }
}

extern "C" void kernel_launch(void* const* d_in, const int* in_sizes, int n_in,
                              void* d_out, int out_size, void* d_ws, size_t ws_size,
                              hipStream_t stream) {
  (void)in_sizes; (void)n_in; (void)out_size; (void)ws_size;
  // setup_inputs order:
  // 0:x 1:H 2:W 3:q_w 4:q_b 5:k_w 6:k_b 7:v_w 8:v_b 9:proj_w 10:proj_b
  // 11:rel_table 12:rel_index
  const float* x  = (const float*)d_in[0];
  const float* qw = (const float*)d_in[3];
  const float* qb = (const float*)d_in[4];
  const float* kw = (const float*)d_in[5];
  const float* kb = (const float*)d_in[6];
  const float* vw = (const float*)d_in[7];
  const float* vb = (const float*)d_in[8];
  const float* pw = (const float*)d_in[9];
  const float* pb = (const float*)d_in[10];
  const float* rt = (const float*)d_in[11];
  const int*   ri = (const int*)d_in[12];
  float* out = (float*)d_out;

  _Float16* wpk = (_Float16*)d_ws;  // 4 * 512*512 f16 = 2 MB

  pack_weights_kernel<<<(WMAT + 255) / 256, 256, 0, stream>>>(qw, kw, vw, pw, wpk);

  const int num_windows = 8192;  // B from the reference
  swin_window_attn_kernel<<<num_windows, 256, 0, stream>>>(
      x, qb, kb, vb, pb, rt, ri, wpk, out);
}
